// LigerFusedLinearGSPOLoss_58669253264202
// MI455X (gfx1250) — matrix operand order repaired and checked by reading, not compiled, not run on hardware
//
#include <hip/hip_runtime.h>
#include <hip/hip_bf16.h>
#include <math.h>
#include <stdint.h>

// Problem sizes (fixed by the reference): B=4, S=1024, H=2048, V=32000
#define GH 2048
#define GV 32000
#define GB 4
#define GS 1024
#define NROWS (GB * GS)       // 4096 token rows
#define MROWS 32              // rows per workgroup (A tile)
#define VSPLIT 2              // V halves per row-tile (grid fill)
#define VHALF (GV / VSPLIT)   // 16000 columns per WG
#define VHTILES (VHALF / 16)  // 1000 16-col tiles per WG
#define NWAVES 8              // 256-thread block = 8 wave32

typedef __attribute__((ext_vector_type(16))) __bf16 v16bf;
typedef __attribute__((ext_vector_type(8)))  float  v8f;
typedef unsigned int u32x4 __attribute__((ext_vector_type(4)));
typedef int          i32x8 __attribute__((ext_vector_type(8)));
typedef int          i32x4 __attribute__((ext_vector_type(4)));

union ABv { v16bf v; unsigned u[8]; };   // 16 bf16 = 8 dwords (A or B fragment)
union ACv { v8f  v; float    f[8]; };    // 16x16 f32 accumulator, 8 VGPRs

#if defined(__has_builtin)
#if __has_builtin(__builtin_amdgcn_tensor_load_to_lds)
#define HAVE_TDM 1
#endif
#endif

// round-to-nearest-even fp32 -> bf16, packed pair (lo in [15:0], hi in [31:16])
__device__ __forceinline__ unsigned bf16pk(float lo, float hi) {
    unsigned a = __float_as_uint(lo);
    unsigned b = __float_as_uint(hi);
    a = (a + 0x7FFFu + ((a >> 16) & 1u)) >> 16;
    b = (b + 0x7FFFu + ((b >> 16) & 1u)) >> 16;
    return a | (b << 16);
}

// online logsumexp fold of one 8-logit accumulator column
__device__ __forceinline__ void lse_fold(float* mx, float* sm, const ACv& acc, float bn) {
#pragma unroll
    for (int r = 0; r < 8; ++r) {
        float v  = acc.f[r] + bn;             // TEMPERATURE == 1
        float nm = fmaxf(mx[r], v);
        sm[r] = sm[r] * __expf(mx[r] - nm) + __expf(v - nm);
        mx[r] = nm;
    }
}

// ---------------------------------------------------------------------------
// Kernel 0: one-time fp32 -> packed bf16 conversion (bandwidth-bound)
// ---------------------------------------------------------------------------
__global__ __launch_bounds__(256)
void cvt_bf16_kernel(const float* __restrict__ src, unsigned* __restrict__ dst,
                     long long n2) {
    long long i = (long long)blockIdx.x * 256 + threadIdx.x;
    const long long stride = (long long)gridDim.x * 256;
    const float2* s2 = (const float2*)src;
    for (; i < n2; i += stride) {
        float2 v = s2[i];
        dst[i] = bf16pk(v.x, v.y);
    }
}

// ---------------------------------------------------------------------------
// Kernel 1: partial (max, sumexp) of (X @ W^T + bias) rows via bf16 WMMA.
// Grid = 128 row-tiles x 2 V-halves. One WG = 32-row A tile (128 KB LDS,
// TDM-staged); each wave owns 32x32 output stripes: per K-step
// 4 ds_load_b128 + 4 global_load_b128 + 4 WMMAs (0.5 KB LDS + 0.5 KB L2 per WMMA).
// ---------------------------------------------------------------------------
__global__ __launch_bounds__(256, 1)
void gspo_lse_kernel(const unsigned* __restrict__ Xbf,   // packed bf16 pairs
                     const unsigned* __restrict__ Wbf,   // packed bf16 pairs
                     const float* __restrict__ bias,
                     float* __restrict__ pmx,            // [VSPLIT][NROWS]
                     float* __restrict__ psm) {          // [VSPLIT][NROWS]
    extern __shared__ __align__(16) unsigned xtile[];    // [MROWS][GH/2] dwords = 128 KB

    const int tid  = threadIdx.x;
    const int lane = tid & 31;
    const int wv   = tid >> 5;
    const int rt   = blockIdx.x & 127;       // row tile
    const int vh   = blockIdx.x >> 7;        // V half
    const int row0 = rt * MROWS;
    const int colb = vh * VHALF;

    // ---- stage the 32 x 2048 bf16 input tile into LDS ----
#ifdef HAVE_TDM
    if (wv == 0) {
        // D# group 0: count=1, lds_addr, 57-bit global byte addr, type=2
        const unsigned ldsAddr = (unsigned)(uintptr_t)xtile;
        const unsigned long long ga =
            (unsigned long long)(uintptr_t)Xbf + (unsigned long long)row0 * GH * 2ull;
        u32x4 g0 = { 1u, ldsAddr,
                     (unsigned)(ga & 0xFFFFFFFFull),
                     (unsigned)((ga >> 32) & 0x01FFFFFFull) | (2u << 30) };
        // D# group 1: data_size=2B; tensor 2048 x 4096, stride0=2048;
        // tile 2048 x 32 (rows contiguous into LDS)
        i32x8 g1 = { (int)(1u << 16),                      // data_size=1 (2B)
                     (int)((GH & 0xFFFFu) << 16),          // tensor_dim0 lo
                     (int)(((unsigned)GH >> 16) | ((NROWS & 0xFFFFu) << 16)),
                     (int)(((unsigned)NROWS >> 16) | ((unsigned)GH << 16)), // tile_dim0
                     (int)MROWS,                           // tile_dim1=32, tile_dim2=0
                     (int)GH,                              // tensor_dim0_stride lo
                     0, 0 };
        i32x4 gz = { 0, 0, 0, 0 };
#if __has_include(<hip/amd_detail/amd_gfx1250_TDM.h>)
        i32x8 gz8 = { 0, 0, 0, 0, 0, 0, 0, 0 };
        __builtin_amdgcn_tensor_load_to_lds(g0, g1, gz, gz, gz8, 0);
#else
        __builtin_amdgcn_tensor_load_to_lds(g0, g1, gz, gz, 0);
#endif
        __builtin_amdgcn_s_wait_tensorcnt(0);
    }
#else
    {
        const unsigned* xin = Xbf + (size_t)row0 * (GH / 2);
        for (int p = tid; p < MROWS * (GH / 2); p += 256) xtile[p] = xin[p];
    }
#endif
    __syncthreads();

    const int nlo  = lane & 15;   // N within tile; also A-row M = lane%16
    const int hi16 = lane >> 4;   // lane group 0/1

    // A fragment LDS bases for the two 16-row fragments
    const unsigned* af0 = xtile + nlo * (GH / 2) + hi16 * 4;
    const unsigned* af1 = af0 + (size_t)16 * (GH / 2);

    // online logsumexp state: [mfrag][colgroup][reg]
    float mx[2][2][8], sm[2][2][8];
#pragma unroll
    for (int f = 0; f < 2; ++f)
#pragma unroll
        for (int c = 0; c < 2; ++c)
#pragma unroll
            for (int r = 0; r < 8; ++r) { mx[f][c][r] = -3.0e38f; sm[f][c][r] = 0.0f; }

    // each wave iteration covers 32 columns x 32 rows (4 WMMA chains)
    for (int tp = wv; tp < VHTILES / 2; tp += NWAVES) {
        const int n0 = colb + tp * 32;
        const unsigned* w0p = Wbf + (size_t)(n0 + nlo) * (GH / 2) + hi16 * 4;
        const unsigned* w1p = w0p + (size_t)16 * (GH / 2);
        if (tp + NWAVES < VHTILES / 2) {
            __builtin_prefetch(Wbf + (size_t)(colb + (tp + NWAVES) * 32 + nlo) * (GH / 2), 0, 1);
        }

        ACv a00, a01, a10, a11;
#pragma unroll
        for (int r = 0; r < 8; ++r) {
            a00.f[r] = 0.0f; a01.f[r] = 0.0f; a10.f[r] = 0.0f; a11.f[r] = 0.0f;
        }

        for (int k0 = 0; k0 < GH; k0 += 32) {
            // A fragments: 2 x 2 ds_load_b128 (each reused by 2 WMMAs)
            ABv fa0, fa1;
            {
                const uint4* ap0 = (const uint4*)(af0 + (k0 >> 1));
                const uint4* ap1 = (const uint4*)(af1 + (k0 >> 1));
                uint4 q0 = ap0[0], q1 = ap0[2], q2 = ap1[0], q3 = ap1[2];
                fa0.u[0] = q0.x; fa0.u[1] = q0.y; fa0.u[2] = q0.z; fa0.u[3] = q0.w;
                fa0.u[4] = q1.x; fa0.u[5] = q1.y; fa0.u[6] = q1.z; fa0.u[7] = q1.w;
                fa1.u[0] = q2.x; fa1.u[1] = q2.y; fa1.u[2] = q2.z; fa1.u[3] = q2.w;
                fa1.u[4] = q3.x; fa1.u[5] = q3.y; fa1.u[6] = q3.z; fa1.u[7] = q3.w;
            }
            // B fragments: 2 x 2 global_load_b128 (each reused by 2 WMMAs)
            ABv fb0, fb1;
            {
                const uint4* b0p = (const uint4*)(w0p + (k0 >> 1));
                const uint4* b1p = (const uint4*)(w1p + (k0 >> 1));
                uint4 r0 = b0p[0], r1 = b0p[2], r2 = b1p[0], r3 = b1p[2];
                fb0.u[0] = r0.x; fb0.u[1] = r0.y; fb0.u[2] = r0.z; fb0.u[3] = r0.w;
                fb0.u[4] = r1.x; fb0.u[5] = r1.y; fb0.u[6] = r1.z; fb0.u[7] = r1.w;
                fb1.u[0] = r2.x; fb1.u[1] = r2.y; fb1.u[2] = r2.z; fb1.u[3] = r2.w;
                fb1.u[4] = r3.x; fb1.u[5] = r3.y; fb1.u[6] = r3.z; fb1.u[7] = r3.w;
            }
            a00.v = __builtin_amdgcn_wmma_f32_16x16x32_bf16(
                        false, fa0.v, false, fb0.v, (short)0, a00.v, false, false);
            a01.v = __builtin_amdgcn_wmma_f32_16x16x32_bf16(
                        false, fa0.v, false, fb1.v, (short)0, a01.v, false, false);
            a10.v = __builtin_amdgcn_wmma_f32_16x16x32_bf16(
                        false, fa1.v, false, fb0.v, (short)0, a10.v, false, false);
            a11.v = __builtin_amdgcn_wmma_f32_16x16x32_bf16(
                        false, fa1.v, false, fb1.v, (short)0, a11.v, false, false);
        }

        const float bn0 = bias[n0 + nlo];
        const float bn1 = bias[n0 + 16 + nlo];
        lse_fold(mx[0][0], sm[0][0], a00, bn0);
        lse_fold(mx[0][1], sm[0][1], a01, bn1);
        lse_fold(mx[1][0], sm[1][0], a10, bn0);
        lse_fold(mx[1][1], sm[1][1], a11, bn1);
    }

    // merge the two column groups per M fragment
#pragma unroll
    for (int f = 0; f < 2; ++f)
#pragma unroll
        for (int r = 0; r < 8; ++r) {
            float nm = fmaxf(mx[f][0][r], mx[f][1][r]);
            sm[f][0][r] = sm[f][0][r] * __expf(mx[f][0][r] - nm)
                        + sm[f][1][r] * __expf(mx[f][1][r] - nm);
            mx[f][0][r] = nm;
        }

    // combine the 16 N-lanes of each lane group
#pragma unroll
    for (int off = 8; off >= 1; off >>= 1) {
#pragma unroll
        for (int f = 0; f < 2; ++f)
#pragma unroll
            for (int r = 0; r < 8; ++r) {
                float omx = __shfl_xor(mx[f][0][r], off, 32);
                float osm = __shfl_xor(sm[f][0][r], off, 32);
                float nm  = fmaxf(mx[f][0][r], omx);
                sm[f][0][r] = sm[f][0][r] * __expf(mx[f][0][r] - nm)
                            + osm * __expf(omx - nm);
                mx[f][0][r] = nm;
            }
    }

    // combine across the 8 waves (reuse xtile LDS after barrier)
    __syncthreads();
    float* red_mx = (float*)xtile;            // [NWAVES][MROWS]
    float* red_sm = red_mx + NWAVES * MROWS;  // [NWAVES][MROWS]
    if (nlo == 0) {
#pragma unroll
        for (int f = 0; f < 2; ++f)
#pragma unroll
            for (int r = 0; r < 8; ++r) {
                int m = f * 16 + r + 8 * hi16;
                red_mx[wv * MROWS + m] = mx[f][0][r];
                red_sm[wv * MROWS + m] = sm[f][0][r];
            }
    }
    __syncthreads();
    if (tid < MROWS) {
        float gm = -3.0e38f, gs = 0.0f;
        for (int w = 0; w < NWAVES; ++w) {
            float omx = red_mx[w * MROWS + tid];
            float osm = red_sm[w * MROWS + tid];
            float nm  = fmaxf(gm, omx);
            gs = gs * __expf(gm - nm) + osm * __expf(omx - nm);
            gm = nm;
        }
        pmx[vh * NROWS + row0 + tid] = gm;
        psm[vh * NROWS + row0 + tid] = gs;
    }
}

// ---------------------------------------------------------------------------
// Kernel 1b: merge the VSPLIT partials into the final logsumexp per row
// ---------------------------------------------------------------------------
__global__ __launch_bounds__(256)
void lse_combine_kernel(const float* __restrict__ pmx,
                        const float* __restrict__ psm,
                        float* __restrict__ lse) {
    int i = blockIdx.x * 256 + threadIdx.x;
    if (i >= NROWS) return;
    float gm = pmx[i], gs = psm[i];
#pragma unroll
    for (int v = 1; v < VSPLIT; ++v) {
        float om = pmx[v * NROWS + i], os = psm[v * NROWS + i];
        float nm = fmaxf(gm, om);
        gs = gs * __expf(gm - nm) + os * __expf(om - nm);
        gm = nm;
    }
    lse[i] = gm + __logf(gs);
}

// ---------------------------------------------------------------------------
// Kernel 2: selected logit per row: dot(X[row], W[token]) + bias[token]
// ---------------------------------------------------------------------------
__global__ __launch_bounds__(256)
void gspo_sel_kernel(const float* __restrict__ X,
                     const float* __restrict__ W,
                     const float* __restrict__ bias,
                     const int* __restrict__ ids,
                     float* __restrict__ sel) {
    __shared__ float wsum[NWAVES];
    const int row = blockIdx.x;
    const int tok = ids[row];
    const float4* x = (const float4*)(X + (size_t)row * GH);
    const float4* w = (const float4*)(W + (size_t)tok * GH);
    float s = 0.0f;
    for (int h = threadIdx.x; h < GH / 4; h += 256) {
        float4 xv = x[h], wv = w[h];
        s += xv.x * wv.x + xv.y * wv.y + xv.z * wv.z + xv.w * wv.w;
    }
    for (int o = 16; o > 0; o >>= 1) s += __shfl_xor(s, o, 32);
    if ((threadIdx.x & 31) == 0) wsum[threadIdx.x >> 5] = s;
    __syncthreads();
    if (threadIdx.x == 0) {
        float t = 0.0f;
        for (int i = 0; i < NWAVES; ++i) t += wsum[i];
        sel[row] = t + bias[tok];
    }
}

// ---------------------------------------------------------------------------
// Kernel 3 helpers / final GSPO loss
// ---------------------------------------------------------------------------
__global__ void gspo_zero_kernel(float* out) { out[0] = 0.0f; }

__global__ __launch_bounds__(256)
void gspo_loss_kernel(const float* __restrict__ lse,
                      const float* __restrict__ sel,
                      const float* __restrict__ mask,
                      const float* __restrict__ adv,
                      const float* __restrict__ ref,
                      const float* __restrict__ oldlp,
                      float* __restrict__ out) {
    __shared__ float w0[NWAVES], w1[NWAVES];
    __shared__ float bc_len, bc_slw;
    const int b = blockIdx.x, tid = threadIdx.x;
    const int lane = tid & 31, wv = tid >> 5;

    // phase 1: seq_len and seq-level log weight
    float sm_m = 0.0f, sm_lr = 0.0f;
    for (int s = tid; s < GS; s += 256) {
        int i = b * GS + s;
        float ptl = sel[i] - lse[i];
        float m   = mask[i];
        sm_m  += m;
        sm_lr += (ptl - oldlp[i]) * m;
    }
    for (int o = 16; o > 0; o >>= 1) {
        sm_m  += __shfl_xor(sm_m, o, 32);
        sm_lr += __shfl_xor(sm_lr, o, 32);
    }
    if (lane == 0) { w0[wv] = sm_m; w1[wv] = sm_lr; }
    __syncthreads();
    if (tid == 0) {
        float tm = 0.0f, tl = 0.0f;
        for (int i = 0; i < NWAVES; ++i) { tm += w0[i]; tl += w1[i]; }
        float seq_len = fmaxf(tm, 1.0f);
        bc_len = seq_len;
        bc_slw = tl / seq_len;
    }
    __syncthreads();

    const float seq_len = bc_len;
    const float slw     = bc_slw;
    // forward: log_importance = ptl - stopgrad(ptl) + slw == slw
    const float c1 = __expf(slw);
    const float c2 = fminf(fmaxf(c1, 1.0f - 0.2f), 1.0f + 0.2f);  // EPS_LOW/HIGH
    const float a  = adv[b];
    const float t1 = -fminf(c1 * a, c2 * a);

    // phase 2: per-token loss with KL penalty
    float sl = 0.0f;
    for (int s = tid; s < GS; s += 256) {
        int i = b * GS + s;
        float ptl = sel[i] - lse[i];
        float d   = ref[i] - ptl;
        float kl  = __expf(d) - d - 1.0f;
        sl += (t1 + 0.04f * kl) * mask[i];     // BETA = 0.04
    }
    for (int o = 16; o > 0; o >>= 1) sl += __shfl_xor(sl, o, 32);
    __syncthreads();                 // protect w0 reuse
    if (lane == 0) w0[wv] = sl;
    __syncthreads();
    if (tid == 0) {
        float t = 0.0f;
        for (int i = 0; i < NWAVES; ++i) t += w0[i];
        atomicAdd(out, (t / seq_len) * (1.0f / (float)GB));
    }
}

// ---------------------------------------------------------------------------
extern "C" void kernel_launch(void* const* d_in, const int* in_sizes, int n_in,
                              void* d_out, int out_size, void* d_ws, size_t ws_size,
                              hipStream_t stream) {
    (void)in_sizes; (void)n_in; (void)out_size; (void)ws_size;
    const float* X     = (const float*)d_in[0];   // (B,S,H) f32
    const float* W     = (const float*)d_in[1];   // (V,H)  f32
    const float* bias  = (const float*)d_in[2];   // (V,)   f32
    const int*   ids   = (const int*)  d_in[3];   // (B,S)  i32
    const float* mask  = (const float*)d_in[4];   // (B,S)  f32
    const float* adv   = (const float*)d_in[5];   // (B,)   f32
    const float* ref   = (const float*)d_in[6];   // (B,S)  f32
    const float* oldlp = (const float*)d_in[7];   // (B,S)  f32
    float* out = (float*)d_out;

    // workspace layout
    float*    lse = (float*)d_ws;                       // [NROWS]
    float*    sel = lse + NROWS;                        // [NROWS]
    float*    pmx = sel + NROWS;                        // [VSPLIT*NROWS]
    float*    psm = pmx + VSPLIT * NROWS;               // [VSPLIT*NROWS]
    unsigned* Xbf = (unsigned*)(psm + VSPLIT * NROWS);  // [NROWS*GH/2] packed bf16
    unsigned* Wbf = Xbf + (size_t)NROWS * (GH / 2);     // [GV*GH/2]    packed bf16

    cvt_bf16_kernel<<<2048, 256, 0, stream>>>(X, Xbf, (long long)NROWS * (GH / 2));
    cvt_bf16_kernel<<<16384, 256, 0, stream>>>(W, Wbf, (long long)GV * (GH / 2));
    gspo_lse_kernel<<<(NROWS / MROWS) * VSPLIT, 256, MROWS * (GH / 2) * 4, stream>>>(
        Xbf, Wbf, bias, pmx, psm);
    lse_combine_kernel<<<NROWS / 256, 256, 0, stream>>>(pmx, psm, lse);
    gspo_sel_kernel<<<NROWS, 256, 0, stream>>>(X, W, bias, ids, sel);
    gspo_zero_kernel<<<1, 1, 0, stream>>>(out);
    gspo_loss_kernel<<<GB, 256, 0, stream>>>(lse, sel, mask, adv, ref, oldlp, out);
}
